// CodeASTEncoder_21182778704863
// MI455X (gfx1250) — matrix operand order repaired and checked by reading, not compiled, not run on hardware
//
#include <hip/hip_runtime.h>
#include <math.h>

typedef _Float16 v16h __attribute__((ext_vector_type(16)));
typedef float    v8f  __attribute__((ext_vector_type(8)));

#define N_NODES 200000
#define N_EDGES 400000
#define VOCAB   32000
#define EMB     256
#define HID     512
#define NG      256
#define H3      (3 * HID)

#define MT_PER_WAVE 25          // 2000 M-tiles / 25 = 80 chunks for the table GEMM
#define POOL_NODES  128         // nodes per pooling block
#define POOL_SLOTS  4           // LDS graph slots per pooling block

// ---------------- WMMA fragment helpers (CDNA5 ISA 7.12.2) -------

// A: 16x32 f16 from row-major fp32. lanes 0-15: M=lane&15, K-half base 0; lanes 16-31: base 16.
// elements 0..15 -> K=kb+0..15. Vectorized float4 loads.
static __device__ __forceinline__ v16h load_a16x32(const float* __restrict__ A,
                                                   int lda, int m0, int k0, int lane) {
  int row = m0 + (lane & 15);
  int kb  = (lane < 16) ? 0 : 16;
  const float4* p = (const float4*)(A + (size_t)row * lda + k0 + kb);
  float4 x0 = p[0], x1 = p[1];        // K = kb+0..7
  float4 y0 = p[2], y1 = p[3];        // K = kb+8..15
  v16h a;
  a[0]  = (_Float16)x0.x; a[1]  = (_Float16)x0.y; a[2]  = (_Float16)x0.z; a[3]  = (_Float16)x0.w;
  a[4]  = (_Float16)x1.x; a[5]  = (_Float16)x1.y; a[6]  = (_Float16)x1.z; a[7]  = (_Float16)x1.w;
  a[8]  = (_Float16)y0.x; a[9]  = (_Float16)y0.y; a[10] = (_Float16)y0.z; a[11] = (_Float16)y0.w;
  a[12] = (_Float16)y1.x; a[13] = (_Float16)y1.y; a[14] = (_Float16)y1.z; a[15] = (_Float16)y1.w;
  return a;
}

// B: 32x16 f16 from row-major B[K,N] (strided column read).
static __device__ __forceinline__ v16h load_b32x16_nn(const float* __restrict__ B,
                                                      int ldb, int k0, int n0, int lane) {
  int col = n0 + (lane & 15);
  int kb  = (lane < 16) ? 0 : 16;
  v16h b;
#pragma unroll
  for (int i = 0; i < 16; ++i)
    b[i] = (_Float16)B[(size_t)(k0 + kb + i) * ldb + col];
  return b;
}

// B^T: 32x16 f16 from row-major W[N,K] (X @ W^T). 16 contiguous floats per lane -> 4x float4.
static __device__ __forceinline__ v16h load_b32x16_nt(const float* __restrict__ W,
                                                      int ldw, int k0, int n0, int lane) {
  int n  = n0 + (lane & 15);
  int kb = (lane < 16) ? 0 : 16;
  const float4* p = (const float4*)(W + (size_t)n * ldw + k0 + kb);
  float4 x0 = p[0], x1 = p[1], x2 = p[2], x3 = p[3];
  v16h b;
  b[0]  = (_Float16)x0.x; b[1]  = (_Float16)x0.y; b[2]  = (_Float16)x0.z; b[3]  = (_Float16)x0.w;
  b[4]  = (_Float16)x1.x; b[5]  = (_Float16)x1.y; b[6]  = (_Float16)x1.z; b[7]  = (_Float16)x1.w;
  b[8]  = (_Float16)x2.x; b[9]  = (_Float16)x2.y; b[10] = (_Float16)x2.z; b[11] = (_Float16)x2.w;
  b[12] = (_Float16)x3.x; b[13] = (_Float16)x3.y; b[14] = (_Float16)x3.z; b[15] = (_Float16)x3.w;
  return b;
}

// C/D f32 16x16: VGPR r -> (M=r, N=lane) lanes 0-15 ; (M=8+r, N=lane-16) lanes 16-31.
static __device__ __forceinline__ void store_c16x16(float* __restrict__ C, int ldc,
                                                    int m0, int n0, int lane, v8f acc) {
  int col   = n0 + (lane & 15);
  int rbase = m0 + ((lane < 16) ? 0 : 8);
#pragma unroll
  for (int r = 0; r < 8; ++r) C[(size_t)(rbase + r) * ldc + col] = acc[r];
}

// ---------------- Table GEMM: twv = emb_table @ gcn_w -----------------------
// B (gcn_w) fragments for the whole K=256 stay register-resident (8 x v16h = 64 VGPRs);
// each wave sweeps 25 M-tiles -> 200 WMMAs per wave, B traffic ~0.
__global__ void k_gemm_nn_breg(const float* __restrict__ A, const float* __restrict__ B,
                               float* __restrict__ C) {
  int wid  = (blockIdx.x * blockDim.x + threadIdx.x) >> 5;
  int lane = threadIdx.x & 31;
  int tn     = wid & 15;        // 16 N-tiles (EMB/16)
  int mchunk = wid >> 4;        // 0..79
  if (mchunk >= (VOCAB / 16) / MT_PER_WAVE) return;  // wave-uniform
  int n0 = tn * 16;

  v16h bf[8];
#pragma unroll
  for (int ks = 0; ks < 8; ++ks) bf[ks] = load_b32x16_nn(B, EMB, ks * 32, n0, lane);

  int arow = (lane & 15);
  int akb  = (lane < 16) ? 0 : 8;
  for (int t = 0; t < MT_PER_WAVE; ++t) {
    int m0 = (mchunk * MT_PER_WAVE + t) * 16;
    v8f acc = {};
#pragma unroll
    for (int ks = 0; ks < 8; ++ks) {
      v16h a = load_a16x32(A, EMB, m0, ks * 32, lane);
      if (t + 1 < MT_PER_WAVE)   // pull next M-tile's A chunk toward L1 (global_prefetch_b8)
        __builtin_prefetch(A + (size_t)(m0 + 16 + arow) * EMB + ks * 32 + akb, 0, 0);
      acc = __builtin_amdgcn_wmma_f32_16x16x32_f16(false, a, false, bf[ks], (short)0, acc,
                                                   false, false);
    }
    store_c16x16(C, EMB, m0, n0, lane, acc);
  }
}

// ---------------- GRU gate GEMM: C[M,N] = A[M,K] @ W[N,K]^T + bias[N] -------
// 16x64 strip per wave: A-fragment reused across 4 WMMAs per k-step.
__global__ void k_gemm_nt4_bias(const float* __restrict__ A, const float* __restrict__ W,
                                const float* __restrict__ bias, float* __restrict__ C,
                                int M, int N, int K) {
  int wid  = (blockIdx.x * blockDim.x + threadIdx.x) >> 5;
  int lane = threadIdx.x & 31;
  int nquads = N >> 6;                    // 64-wide strips
  int tm = wid / nquads, tq = wid % nquads;
  if (tm * 16 >= M) return;               // wave-uniform
  int m0 = tm * 16;
  int nb = tq * 64;

  v8f acc[4] = {};
  for (int k0 = 0; k0 < K; k0 += 32) {
    v16h a = load_a16x32(A, K, m0, k0, lane);
#pragma unroll
    for (int q = 0; q < 4; ++q) {
      v16h b = load_b32x16_nt(W, K, k0, nb + q * 16, lane);
      acc[q] = __builtin_amdgcn_wmma_f32_16x16x32_f16(false, a, false, b, (short)0, acc[q],
                                                      false, false);
    }
  }
#pragma unroll
  for (int q = 0; q < 4; ++q) {
    float bi = bias[nb + q * 16 + (lane & 15)];
#pragma unroll
    for (int r = 0; r < 8; ++r) acc[q][r] += bi;
    store_c16x16(C, N, m0, nb + q * 16, lane, acc[q]);
  }
}

// ---------------- Graph kernels --------------------------------------------

__global__ void k_deg_init(float* __restrict__ dinv) {          // self-loop = 1
  int i = blockIdx.x * blockDim.x + threadIdx.x;
  if (i < N_NODES) dinv[i] = 1.0f;
}

__global__ void k_deg_edges(const int* __restrict__ e_dst, float* __restrict__ dinv) {
  int j = blockIdx.x * blockDim.x + threadIdx.x;
  if (j < N_EDGES) atomicAdd(&dinv[e_dst[j]], 1.0f);
}

__global__ void k_deg_rsqrt(float* __restrict__ dinv) {
  int i = blockIdx.x * blockDim.x + threadIdx.x;
  if (i < N_NODES) dinv[i] = rsqrtf(fmaxf(dinv[i], 1.0f));
}

// ctx[i] = gcn_b + self-loop contribution (plain stores)
__global__ void k_ctx_init(const int* __restrict__ v, const float* __restrict__ twv,
                           const float* __restrict__ dinv, const float* __restrict__ gcn_b,
                           float* __restrict__ ctx) {
  int i = blockIdx.x, f = threadIdx.x;
  float di = dinv[i];
  ctx[(size_t)i * EMB + f] = gcn_b[f] + twv[(size_t)v[i] * EMB + f] * di * di;
}

// message scatter: ctx[dst] += twv[v[src]] * norm  (twv gathers hit the 192MB L2)
__global__ void k_scatter(const int* __restrict__ e_src, const int* __restrict__ e_dst,
                          const int* __restrict__ v, const float* __restrict__ twv,
                          const float* __restrict__ dinv, float* __restrict__ ctx) {
  int j = blockIdx.x, f = threadIdx.x;
  int s = e_src[j], d = e_dst[j];
  float nm = dinv[s] * dinv[d];
  atomicAdd(&ctx[(size_t)d * EMB + f], twv[(size_t)v[s] * EMB + f] * nm);
}

// ---------------- Pooling ---------------------------------------------------

static __device__ __forceinline__ void atomicMaxF(float* addr, float val) {
  if (val >= 0.0f) atomicMax((int*)addr, __float_as_int(val));
  else             atomicMin((unsigned int*)addr, __float_as_uint(val));
}

__global__ void k_pool_init(float* __restrict__ mean_p, float* __restrict__ max_p,
                            float* __restrict__ cnt) {
  int i = blockIdx.x * blockDim.x + threadIdx.x;
  if (i < NG * EMB) { mean_p[i] = 0.0f; max_p[i] = -INFINITY; }
  if (i < NG) cnt[i] = 0.0f;
}

// batch_ind is sorted: a 128-node block spans ~1-2 graphs. Accumulate sum/max in
// LDS slots (plain per-lane LDS ops), flush once per block; fallback atomics for
// the rare block spanning >POOL_SLOTS graphs.
__global__ void k_pool(const int* __restrict__ batch_ind, const float* __restrict__ ctx,
                       float* __restrict__ mean_p, float* __restrict__ max_p,
                       float* __restrict__ cnt) {
  __shared__ float s_sum[POOL_SLOTS][EMB];
  __shared__ float s_max[POOL_SLOTS][EMB];
  __shared__ float s_cnt[POOL_SLOTS];
  int f    = threadIdx.x;
  int base = blockIdx.x * POOL_NODES;
#pragma unroll
  for (int s = 0; s < POOL_SLOTS; ++s) { s_sum[s][f] = 0.0f; s_max[s][f] = -INFINITY; }
  if (f < POOL_SLOTS) s_cnt[f] = 0.0f;
  __syncthreads();

  int gbase = batch_ind[base];                 // min graph id in this block (sorted)
  for (int t = 0; t < POOL_NODES; ++t) {
    int i = base + t;
    if (i >= N_NODES) break;                   // uniform across block
    int g = batch_ind[i];                      // uniform across block
    float val = ctx[(size_t)i * EMB + f];
    int s = g - gbase;
    if (s < POOL_SLOTS) {                      // lane f owns [s][f]: no races
      s_sum[s][f] += val;
      s_max[s][f] = fmaxf(s_max[s][f], val);
      if (f == 0) s_cnt[s] += 1.0f;
    } else {                                   // rare fallback
      atomicAdd(&mean_p[(size_t)g * EMB + f], val);
      atomicMaxF(&max_p[(size_t)g * EMB + f], val);
      if (f == 0) atomicAdd(&cnt[g], 1.0f);
    }
  }
  __syncthreads();
#pragma unroll
  for (int s = 0; s < POOL_SLOTS; ++s) {
    if (s_cnt[s] > 0.0f) {
      int g = gbase + s;
      atomicAdd(&mean_p[(size_t)g * EMB + f], s_sum[s][f]);
      atomicMaxF(&max_p[(size_t)g * EMB + f], s_max[s][f]);
      if (f == 0) atomicAdd(&cnt[g], s_cnt[s]);
    }
  }
}

__global__ void k_pool_fin(float* __restrict__ mean_p, float* __restrict__ max_p,
                           const float* __restrict__ cnt) {
  int i = blockIdx.x * blockDim.x + threadIdx.x;
  if (i >= NG * EMB) return;
  int g = i / EMB;
  mean_p[i] /= fmaxf(cnt[g], 1.0f);
  float m = max_p[i];
  max_p[i] = (m > -INFINITY) ? m : 0.0f;       // isfinite -> else 0
}

// ---------------- GRU -------------------------------------------------------

static __device__ __forceinline__ float sigm(float x) { return 1.0f / (1.0f + expf(-x)); }

__global__ void k_gru1(const float* __restrict__ gi, const float* __restrict__ b_hh,
                       float* __restrict__ h1) {
  int idx = blockIdx.x * blockDim.x + threadIdx.x;
  if (idx >= NG * HID) return;
  int b = idx / HID, h = idx % HID;
  const float* g = gi + (size_t)b * H3;
  float r = sigm(g[h] + b_hh[h]);
  float z = sigm(g[HID + h] + b_hh[HID + h]);
  float n = tanhf(g[2 * HID + h] + r * b_hh[2 * HID + h]);
  h1[idx] = (1.0f - z) * n;
}

__global__ void k_gru2(const float* __restrict__ gi, const float* __restrict__ gh,
                       const float* __restrict__ h1, float* __restrict__ out_a,
                       float* __restrict__ out_b) {
  int idx = blockIdx.x * blockDim.x + threadIdx.x;
  if (idx >= NG * HID) return;
  int b = idx / HID, h = idx % HID;
  const float* gix = gi + (size_t)b * H3;
  const float* ghx = gh + (size_t)b * H3;
  float r = sigm(gix[h] + ghx[h]);
  float z = sigm(gix[HID + h] + ghx[HID + h]);
  float n = tanhf(gix[2 * HID + h] + r * ghx[2 * HID + h]);
  float hn = (1.0f - z) * n + z * h1[idx];
  out_a[idx] = hn;
  out_b[idx] = hn;
}

// ---------------- Launcher --------------------------------------------------

extern "C" void kernel_launch(void* const* d_in, const int* in_sizes, int n_in,
                              void* d_out, int out_size, void* d_ws, size_t ws_size,
                              hipStream_t stream) {
  const int*   v     = (const int*)d_in[0];
  const int*   e     = (const int*)d_in[1];     // [2, N_EDGES] row-major
  const int*   batch = (const int*)d_in[2];
  const float* emb   = (const float*)d_in[3];   // [VOCAB, EMB]
  const float* gcn_w = (const float*)d_in[4];   // [EMB, EMB]
  const float* gcn_b = (const float*)d_in[5];   // [EMB]
  const float* w_ih  = (const float*)d_in[6];   // [3H, EMB]
  const float* w_hh  = (const float*)d_in[7];   // [3H, HID]
  const float* b_ih  = (const float*)d_in[8];   // [3H]
  const float* b_hh  = (const float*)d_in[9];   // [3H]
  (void)in_sizes; (void)n_in; (void)out_size; (void)ws_size;

  const int* e_src = e;
  const int* e_dst = e + N_EDGES;

  float* ws = (float*)d_ws;
  size_t o = 0;
  float* ctx    = ws + o; o += (size_t)N_NODES * EMB;   // 204.8 MB
  float* twv    = ws + o; o += (size_t)VOCAB * EMB;     // 32.8 MB (L2-resident)
  float* dinv   = ws + o; o += N_NODES;
  float* cnt    = ws + o; o += NG;
  float* mean_p = ws + o; o += (size_t)NG * EMB;
  float* max_p  = ws + o; o += (size_t)NG * EMB;
  float* gi     = ws + o; o += (size_t)NG * H3;
  float* gh     = ws + o; o += (size_t)NG * H3;

  float* out = (float*)d_out;
  float* h1  = out;                        // outs[0]
  float* h2a = out + (size_t)NG * HID;     // outs[1]
  float* h2b = out + (size_t)2 * NG * HID; // h_last

  // 1) twv = emb_table @ gcn_w : 1280 waves, B register-resident, 200 WMMA/wave
  k_gemm_nn_breg<<<160, 256, 0, stream>>>(emb, gcn_w, twv);
  // 2) degrees -> dinv
  k_deg_init <<<(N_NODES + 255) / 256, 256, 0, stream>>>(dinv);
  k_deg_edges<<<(N_EDGES + 255) / 256, 256, 0, stream>>>(e_dst, dinv);
  k_deg_rsqrt<<<(N_NODES + 255) / 256, 256, 0, stream>>>(dinv);
  // 3) ctx = gcn_b + self-loop ; edge scatter
  k_ctx_init<<<N_NODES, EMB, 0, stream>>>(v, twv, dinv, gcn_b, ctx);
  k_scatter <<<N_EDGES, EMB, 0, stream>>>(e_src, e_dst, v, twv, dinv, ctx);
  // 4) per-graph mean/max pooling (LDS two-stage)
  k_pool_init<<<(NG * EMB + 255) / 256, 256, 0, stream>>>(mean_p, max_p, cnt);
  k_pool     <<<(N_NODES + POOL_NODES - 1) / POOL_NODES, EMB, 0, stream>>>(batch, ctx,
                                                                           mean_p, max_p, cnt);
  k_pool_fin <<<(NG * EMB + 255) / 256, 256, 0, stream>>>(mean_p, max_p, cnt);
  // 5) GRU step 1: gi = mean_p @ w_ih^T + b_ih ; h1 = (1-z)*n
  {
    int waves = (NG / 16) * (H3 / 64);   // 384 waves
    k_gemm_nt4_bias<<<waves * 32 / 256, 256, 0, stream>>>(mean_p, w_ih, b_ih, gi, NG, H3, EMB);
    k_gru1<<<(NG * HID + 255) / 256, 256, 0, stream>>>(gi, b_hh, h1);
  }
  // 6) GRU step 2
  {
    int waves = (NG / 16) * (H3 / 64);
    k_gemm_nt4_bias<<<waves * 32 / 256, 256, 0, stream>>>(max_p, w_ih, b_ih, gi, NG, H3, EMB);
    k_gemm_nt4_bias<<<waves * 32 / 256, 256, 0, stream>>>(h1, w_hh, b_hh, gh, NG, H3, HID);
    k_gru2<<<(NG * HID + 255) / 256, 256, 0, stream>>>(gi, gh, h1, h2a, h2b);
  }
}
